// AttentionWithPairBias_67259187855461
// MI455X (gfx1250) — matrix operand order repaired and checked by reading, not compiled, not run on hardware
//
#include <hip/hip_runtime.h>
#include <math.h>

typedef __attribute__((ext_vector_type(16))) _Float16 v16h;
typedef __attribute__((ext_vector_type(8)))  float    v8f;
typedef __attribute__((ext_vector_type(4)))  int      v4i;
typedef __attribute__((ext_vector_type(8)))  int      v8i;

#define LSEQ 1024
#define CS   768
#define CZ   64
#define NH   12
#define HD   64

#define USE_TDM 1

struct F16x16 { union { v16h v; uint4 q[2]; }; };

// ---------------------------------------------------------------------------
// WMMA fragment loaders (wave32; k = 16*h + s in both A and B, this file's own order)
// ---------------------------------------------------------------------------

// A fragment (16x32 f16) from row-major [*, ld] at (row0, k0): 2x b128.
__device__ inline v16h load_A(const _Float16* p, int ld, int row0, int k0) {
  int lane = threadIdx.x & 31;
  int r = lane & 15, hl = lane >> 4;
  const _Float16* base = p + (size_t)(row0 + r) * ld + k0 + hl * 16;
  F16x16 A;
  A.q[0] = *(const uint4*)(base);        // K offsets 0..7
  A.q[1] = *(const uint4*)(base + 8);    // K offsets 8..15
  return A.v;
}

// Packed-B layout: fragment (k0/32, n0/16) stored as [lane][16 halves].
__device__ inline size_t packB_index(int K, int N, int Ntot) {
  int lane = (((K >> 4) & 1) << 4) | (N & 15);
  return ((size_t)((K >> 5) * (Ntot >> 4) + (N >> 4)) * 32 + lane) * 16 + (K & 15);
}

__device__ inline unsigned packh2(float lo, float hi) {
  union { _Float16 h[2]; unsigned u; } p;
  p.h[0] = (_Float16)lo; p.h[1] = (_Float16)hi;
  return p.u;
}

// ---------------------------------------------------------------------------
// LayerNorm(single) -> f16   (one wave per row of 768; row cached in regs)
// ---------------------------------------------------------------------------
__global__ void ln_single_kernel(const float* __restrict__ x,
                                 const float* __restrict__ g,
                                 const float* __restrict__ b,
                                 _Float16* __restrict__ out) {
  int wave = (blockIdx.x * blockDim.x + threadIdx.x) >> 5;
  int lane = threadIdx.x & 31;
  if (wave >= LSEQ) return;
  const float* row = x + (size_t)wave * CS;
  float vals[CS / 32];
  float s = 0.f, s2 = 0.f;
#pragma unroll
  for (int c = 0; c < CS / 32; ++c) {
    float v = row[lane + 32 * c];
    vals[c] = v; s += v; s2 += v * v;
  }
#pragma unroll
  for (int m = 16; m >= 1; m >>= 1) { s += __shfl_xor(s, m, 32); s2 += __shfl_xor(s2, m, 32); }
  float mu = s * (1.f / CS);
  float var = s2 * (1.f / CS) - mu * mu;
  float rs = rsqrtf(var + 1e-5f);
  _Float16* o = out + (size_t)wave * CS;
#pragma unroll
  for (int c = 0; c < CS / 32; ++c) {
    int i = lane + 32 * c;
    o[i] = (_Float16)((vals[c] - mu) * rs * g[i] + b[i]);
  }
}

// ---------------------------------------------------------------------------
// Pack kernels: scatter source into WMMA B-fragment layout (one-time cost)
// ---------------------------------------------------------------------------
__global__ void pack_w_kernel(const float* __restrict__ in, _Float16* __restrict__ out,
                              int K, int Ntot) {
  int i = blockIdx.x * blockDim.x + threadIdx.x;
  if (i >= K * Ntot) return;
  int kk = i / Ntot, n = i - kk * Ntot;
  out[packB_index(kk, n, Ntot)] = (_Float16)in[i];
}

__global__ void pack_h_kernel(const _Float16* __restrict__ in, _Float16* __restrict__ out,
                              int K, int Ntot) {
  int i = blockIdx.x * blockDim.x + threadIdx.x;
  if (i >= K * Ntot) return;
  int kk = i / Ntot, n = i - kk * Ntot;
  out[packB_index(kk, n, Ntot)] = in[i];
}

// ---------------------------------------------------------------------------
// WMMA GEMM, explicit double-buffered pipeline (no rotation copies).
// One wave -> 16x64 tile. K must be a multiple of 64.
// ---------------------------------------------------------------------------
__device__ __forceinline__ void gemm_load(F16x16& a, F16x16* b,
                                          const _Float16* ap, const _Float16* bp) {
  a.q[0] = *(const uint4*)(ap);
  a.q[1] = *(const uint4*)(ap + 8);
#pragma unroll
  for (int t = 0; t < 4; ++t) {
    b[t].q[0] = *(const uint4*)(bp + t * 512);
    b[t].q[1] = *(const uint4*)(bp + t * 512 + 8);
  }
}

__global__ void gemm_kernel(const _Float16* __restrict__ A, const _Float16* __restrict__ Bp,
                            const float* __restrict__ bias,
                            float* __restrict__ Cf, _Float16* __restrict__ Ch,
                            int M, int N, int K, float scale) {
  int wave = (blockIdx.x * blockDim.x + threadIdx.x) >> 5;
  int ntile = N >> 6;
  int m0 = (wave / ntile) * 16;
  int n0 = (wave % ntile) * 64;
  if (m0 >= M) return;
  int lane = threadIdx.x & 31;
  int r = lane & 15, hl = lane >> 4;

  const _Float16* ap = A + (size_t)(m0 + r) * K + hl * 16;
  const _Float16* bp = Bp + ((size_t)(n0 >> 4) * 32 + lane) * 16;
  const size_t bstep = (size_t)(N >> 4) * 512;   // halves per 32-wide K step

  F16x16 a, b[4], a2, b2[4];
  gemm_load(a, b, ap, bp);

  v8f acc[4] = {};
  for (int k0 = 0; k0 < K; k0 += 64) {
    ap += 32; bp += bstep;
    gemm_load(a2, b2, ap, bp);                   // k0+32 (K%64==0 -> in range)
#pragma unroll
    for (int t = 0; t < 4; ++t)
      acc[t] = __builtin_amdgcn_wmma_f32_16x16x32_f16(false, a.v, false, b[t].v,
                                                      (short)0, acc[t], false, false);
    ap += 32; bp += bstep;
    if (k0 + 64 < K) gemm_load(a, b, ap, bp);    // k0+64
#pragma unroll
    for (int t = 0; t < 4; ++t)
      acc[t] = __builtin_amdgcn_wmma_f32_16x16x32_f16(false, a2.v, false, b2[t].v,
                                                      (short)0, acc[t], false, false);
  }

#pragma unroll
  for (int t = 0; t < 4; ++t) {
    int col = n0 + 16 * t + r;
    float bb = bias ? bias[col] : 0.f;
#pragma unroll
    for (int rr = 0; rr < 8; ++rr) {
      int row = m0 + rr + 8 * hl;
      float v = acc[t][rr] * scale + bb;
      if (Cf) Cf[(size_t)row * N + col] = v;
      if (Ch) Ch[(size_t)row * N + col] = (_Float16)v;
    }
  }
}

// ---------------------------------------------------------------------------
// Fused pair pass: double-buffered TDM DMA of 16x64 f32 tiles into LDS,
// LN over C_Z=64, then zn @ Wb via WMMA. Single read of the 256 MB tensor.
// ---------------------------------------------------------------------------
#if USE_TDM
__device__ inline void tdm_load_tile(const float* src, void* lds_dst) {
  // D# per CDNA5 ISA: 2D tile, 4B elements, 64x16 tile.
  unsigned lds_off = (unsigned)(size_t)lds_dst;           // aperture low 32b = LDS offset
  unsigned long long ga = (unsigned long long)(size_t)src;
  v4i g0;
  g0[0] = 1;                                              // count=1, is_restore=0
  g0[1] = (int)lds_off;                                   // lds_addr
  g0[2] = (int)(ga & 0xffffffffu);                        // global_addr[31:0]
  g0[3] = (int)((ga >> 32) & 0x1ffffffu) | (2 << 30);     // global_addr[56:32], type=2
  v8i g1;
  g1[0] = 2 << 16;                 // workgroup_mask=0, data_size=2 (4 bytes)
  g1[1] = 64 << 16;                // tensor_dim0[15:0]=64
  g1[2] = 0;
  g1[3] = 0x10 | (64 << 16);       // tensor_dim1[31:16]=0x10 (1M rows); tile_dim0=64
  g1[4] = 16;                      // tile_dim1=16
  g1[5] = 64;                      // tensor_dim0_stride=64 elements
  g1[6] = 0;
  g1[7] = 0;
  asm volatile("tensor_load_to_lds %0, %1" :: "s"(g0), "s"(g1) : "memory");
}
#endif

__global__ void pair_bias_kernel(const float* __restrict__ pair,
                                 const float* __restrict__ g,
                                 const float* __restrict__ b,
                                 const float* __restrict__ Wb,
                                 _Float16* __restrict__ bias) {
  __shared__ __align__(16) float    zf[8][2][16 * 64];
  __shared__ __align__(16) _Float16 zh[8][16 * 64];
  int wv = threadIdx.x >> 5, lane = threadIdx.x & 31;
  int r = lane & 15, hl = lane >> 4;
  const int NT = 4;                                       // tiles per wave
  long wave = (long)blockIdx.x * 8 + wv;
  long t0base = wave * NT * 16;

  // B fragments for Wb (64x12 padded to 16 cols), built once
  F16x16 B0, B1;
  {
    union { v16h v; _Float16 h[16]; } tb0, tb1;
#pragma unroll
    for (int s = 0; s < 16; ++s) {
      int kk = hl * 16 + s;
      tb0.h[s] = (r < NH) ? (_Float16)Wb[kk * NH + r]        : (_Float16)0.f;
      tb1.h[s] = (r < NH) ? (_Float16)Wb[(kk + 32) * NH + r] : (_Float16)0.f;
    }
    B0.v = tb0.v; B1.v = tb1.v;
  }

  // per-lane gamma/beta for channels hl*32 .. hl*32+31
  float4 g4[8], b4[8];
#pragma unroll
  for (int c = 0; c < 8; ++c) {
    g4[c] = ((const float4*)g)[hl * 8 + c];
    b4[c] = ((const float4*)b)[hl * 8 + c];
  }

#if USE_TDM
  tdm_load_tile(pair + t0base * 64, &zf[wv][0][0]);
#endif

  for (int tt = 0; tt < NT; ++tt) {
    long t0 = t0base + (long)tt * 16;
#if USE_TDM
    if (tt + 1 < NT) {
      tdm_load_tile(pair + (t0 + 16) * 64, &zf[wv][(tt + 1) & 1][0]);
      __builtin_amdgcn_s_wait_tensorcnt(1);
    } else {
      __builtin_amdgcn_s_wait_tensorcnt(0);
    }
    float* lf = &zf[wv][tt & 1][0];
#else
    float* lf = &zf[wv][tt & 1][0];
    {
      const float* src = pair + t0 * 64;
      for (int i = lane; i < 256; i += 32)
        ((float4*)lf)[i] = ((const float4*)src)[i];
      asm volatile("s_wait_dscnt 0" ::: "memory");
    }
#endif
    // stats: row r split across lanes r (ch 0..31) and r+16 (ch 32..63)
    const float4* rowp = (const float4*)(lf + r * 64 + hl * 32);
    float4 v4[8];
    float s = 0.f, s2 = 0.f;
#pragma unroll
    for (int c = 0; c < 8; ++c) {
      v4[c] = rowp[c];
      s  += v4[c].x + v4[c].y + v4[c].z + v4[c].w;
      s2 += v4[c].x * v4[c].x + v4[c].y * v4[c].y + v4[c].z * v4[c].z + v4[c].w * v4[c].w;
    }
    s += __shfl_xor(s, 16, 32); s2 += __shfl_xor(s2, 16, 32);
    float mu = s * (1.f / 64.f);
    float var = s2 * (1.f / 64.f) - mu * mu;
    float rs = rsqrtf(var + 1e-5f);

    unsigned* lhw = (unsigned*)(zh[wv] + r * 64 + hl * 32);
#pragma unroll
    for (int c = 0; c < 8; ++c) {
      float x0 = (v4[c].x - mu) * rs * g4[c].x + b4[c].x;
      float x1 = (v4[c].y - mu) * rs * g4[c].y + b4[c].y;
      float x2 = (v4[c].z - mu) * rs * g4[c].z + b4[c].z;
      float x3 = (v4[c].w - mu) * rs * g4[c].w + b4[c].w;
      lhw[c * 2 + 0] = packh2(x0, x1);
      lhw[c * 2 + 1] = packh2(x2, x3);
    }
    asm volatile("s_wait_dscnt 0" ::: "memory");

    v16h a0 = load_A(zh[wv], 64, 0, 0);
    v16h a1 = load_A(zh[wv], 64, 0, 32);
    v8f acc = {};
    acc = __builtin_amdgcn_wmma_f32_16x16x32_f16(false, a0, false, B0.v, (short)0, acc, false, false);
    acc = __builtin_amdgcn_wmma_f32_16x16x32_f16(false, a1, false, B1.v, (short)0, acc, false, false);

    // one b128 store per lane: 8 contiguous halves of head plane r
    if (r < NH) {
      union { _Float16 h[8]; uint4 qv; } st;
#pragma unroll
      for (int rr = 0; rr < 8; ++rr) st.h[rr] = (_Float16)acc[rr];
      *(uint4*)(bias + (size_t)r * ((size_t)LSEQ * LSEQ) + t0 + 8 * hl) = st.qv;
    }
  }
}

// ---------------------------------------------------------------------------
// Flash attention: one wave per (head, 16-row i-tile). Explicit double-buffer
// of k-fragments across j-steps (no rotation copies); v pre-packed; q scaled.
// ---------------------------------------------------------------------------
__device__ __forceinline__ void loadK4(F16x16* kb, const _Float16* kp) {
#pragma unroll
  for (int t = 0; t < 2; ++t) {
    kb[2 * t].q[0]     = *(const uint4*)(kp + (size_t)t * 16 * CS);
    kb[2 * t].q[1]     = *(const uint4*)(kp + (size_t)t * 16 * CS + 8);
    kb[2 * t + 1].q[0] = *(const uint4*)(kp + (size_t)t * 16 * CS + 32);
    kb[2 * t + 1].q[1] = *(const uint4*)(kp + (size_t)t * 16 * CS + 40);
  }
}

__device__ __forceinline__ void attn_step(
    bool prefetchNext,
    const _Float16*& kp, const _Float16*& vpp, const _Float16*& bp2,
    F16x16* kuse, F16x16* kpre,
    const v16h& qa0, const v16h& qa1,
    v8f* Oacc, float* m, float* l, _Float16* pb, int n, int hl) {
  // v fragments + bias values for this step (consumed late)
  F16x16 vb[4];
#pragma unroll
  for (int t = 0; t < 4; ++t) {
    vb[t].q[0] = *(const uint4*)(vpp + t * 512);
    vb[t].q[1] = *(const uint4*)(vpp + t * 512 + 8);
  }
  _Float16 bv[16];
#pragma unroll
  for (int t = 0; t < 2; ++t)
#pragma unroll
    for (int rr = 0; rr < 8; ++rr)
      bv[t * 8 + rr] = bp2[rr * LSEQ + 16 * t];
  vpp += (size_t)48 * 512;
  bp2 += 32;

  // logits
  v8f Lt[2];
#pragma unroll
  for (int t = 0; t < 2; ++t) {
    v8f acc = {};
    acc = __builtin_amdgcn_wmma_f32_16x16x32_f16(false, qa0, false, kuse[2 * t].v,
                                                 (short)0, acc, false, false);
    acc = __builtin_amdgcn_wmma_f32_16x16x32_f16(false, qa1, false, kuse[2 * t + 1].v,
                                                 (short)0, acc, false, false);
#pragma unroll
    for (int rr = 0; rr < 8; ++rr) acc[rr] += (float)bv[t * 8 + rr];
    Lt[t] = acc;
  }

  // prefetch next j-step's k fragments; softmax below hides the latency
  kp += (size_t)32 * CS;
  if (prefetchNext) loadK4(kpre, kp);

  // online softmax per row
#pragma unroll
  for (int r = 0; r < 8; ++r) {
    float mm = fmaxf(Lt[0][r], Lt[1][r]);
#pragma unroll
    for (int s = 8; s >= 1; s >>= 1) mm = fmaxf(mm, __shfl_xor(mm, s, 32));
    float mn = fmaxf(m[r], mm);
    float alpha = __expf(m[r] - mn);
    m[r] = mn;
    float p0 = __expf(Lt[0][r] - mn);
    float p1 = __expf(Lt[1][r] - mn);
    float rsum = p0 + p1;
#pragma unroll
    for (int s = 8; s >= 1; s >>= 1) rsum += __shfl_xor(rsum, s, 32);
    l[r] = l[r] * alpha + rsum;
#pragma unroll
    for (int t = 0; t < 4; ++t) Oacc[t][r] *= alpha;
    pb[(r + 8 * hl) * 32 + n]      = (_Float16)p0;
    pb[(r + 8 * hl) * 32 + 16 + n] = (_Float16)p1;
  }
  asm volatile("s_wait_dscnt 0" ::: "memory");
  v16h pa = load_A(pb, 32, 0, 0);
#pragma unroll
  for (int t = 0; t < 4; ++t)
    Oacc[t] = __builtin_amdgcn_wmma_f32_16x16x32_f16(false, pa, false, vb[t].v,
                                                     (short)0, Oacc[t], false, false);
}

__global__ void attn_kernel(const _Float16* __restrict__ q,
                            const _Float16* __restrict__ k,
                            const _Float16* __restrict__ vp,
                            const _Float16* __restrict__ bias,
                            _Float16* __restrict__ o) {
  __shared__ __align__(16) _Float16 pbuf[4][16 * 32];
  int wv = threadIdx.x >> 5, lane = threadIdx.x & 31;
  int wave = blockIdx.x * 4 + wv;
  int h  = wave >> 6;
  int i0 = (wave & 63) * 16;
  int n = lane & 15, hl = lane >> 4;

  v16h qa0 = load_A(q, CS, i0, h * HD);
  v16h qa1 = load_A(q, CS, i0, h * HD + 32);

  v8f Oacc[4] = {};
  float m[8], l[8];
#pragma unroll
  for (int r = 0; r < 8; ++r) { m[r] = -1e30f; l[r] = 0.f; }

  const _Float16* kp  = k + (size_t)n * CS + h * HD + hl * 16;
  const _Float16* vpp = vp + ((size_t)(h * 4) * 32 + lane) * 16;
  const _Float16* bp2 = bias + (long)h * LSEQ * LSEQ + (size_t)(i0 + 8 * hl) * LSEQ + n;
  _Float16* pb = pbuf[wv];

  F16x16 kcur[4], knxt[4];
  loadK4(kcur, kp);

  for (int j0 = 0; j0 < LSEQ; j0 += 64) {
    attn_step(true,              kp, vpp, bp2, kcur, knxt, qa0, qa1, Oacc, m, l, pb, n, hl);
    attn_step(j0 + 64 < LSEQ,    kp, vpp, bp2, knxt, kcur, qa0, qa1, Oacc, m, l, pb, n, hl);
  }

#pragma unroll
  for (int t = 0; t < 4; ++t)
#pragma unroll
    for (int r = 0; r < 8; ++r) {
      float val = Oacc[t][r] / l[r];
      o[(size_t)(i0 + r + 8 * hl) * CS + h * HD + 16 * t + n] = (_Float16)val;
    }
}

// ---------------------------------------------------------------------------
// out = single + sigmoid(gate) * oproj
// ---------------------------------------------------------------------------
__global__ void final_kernel(const float* __restrict__ single,
                             const float* __restrict__ gate,
                             const float* __restrict__ oproj,
                             float* __restrict__ out, int n) {
  int i = blockIdx.x * blockDim.x + threadIdx.x;
  if (i < n) {
    float s = 1.f / (1.f + __expf(-gate[i]));
    out[i] = single[i] + s * oproj[i];
  }
}

// ---------------------------------------------------------------------------
// Host orchestration
// ---------------------------------------------------------------------------
static inline size_t align256(size_t x) { return (x + 255) & ~(size_t)255; }

extern "C" void kernel_launch(void* const* d_in, const int* in_sizes, int n_in,
                              void* d_out, int out_size, void* d_ws, size_t ws_size,
                              hipStream_t stream) {
  (void)in_sizes; (void)n_in; (void)out_size; (void)ws_size;
  const float* single = (const float*)d_in[0];
  const float* pair   = (const float*)d_in[1];
  const float* ln_s_g = (const float*)d_in[2];
  const float* ln_s_b = (const float*)d_in[3];
  const float* ln_z_g = (const float*)d_in[4];
  const float* ln_z_b = (const float*)d_in[5];
  const float* Wq     = (const float*)d_in[6];
  const float* Wk     = (const float*)d_in[7];
  const float* Wv     = (const float*)d_in[8];
  const float* Wb     = (const float*)d_in[9];
  const float* Wout   = (const float*)d_in[10];
  const float* bout   = (const float*)d_in[11];
  const float* Wgate  = (const float*)d_in[12];
  const float* bgate  = (const float*)d_in[13];
  float* out = (float*)d_out;

  char* w = (char*)d_ws;
  size_t off = 0;
  auto alloc = [&](size_t bytes) { void* p = w + off; off = align256(off + bytes); return p; };

  const size_t SC = (size_t)LSEQ * CS;
  const size_t WW = (size_t)CS * CS;
  _Float16* s16    = (_Float16*)alloc(SC * 2);
  _Float16* WqP    = (_Float16*)alloc(WW * 2);
  _Float16* WkP    = (_Float16*)alloc(WW * 2);
  _Float16* WvP    = (_Float16*)alloc(WW * 2);
  _Float16* WgP    = (_Float16*)alloc(WW * 2);
  _Float16* WoP    = (_Float16*)alloc(WW * 2);
  _Float16* q16    = (_Float16*)alloc(SC * 2);
  _Float16* k16    = (_Float16*)alloc(SC * 2);
  _Float16* v16    = (_Float16*)alloc(SC * 2);
  _Float16* vpack  = (_Float16*)alloc(SC * 2);
  _Float16* o16    = (_Float16*)alloc(SC * 2);
  float*    gate   = (float*)alloc(SC * 4);
  float*    oproj  = (float*)alloc(SC * 4);
  _Float16* bias16 = (_Float16*)alloc((size_t)NH * LSEQ * LSEQ * 2);

  // 1) LN(single) -> s16
  ln_single_kernel<<<128, 256, 0, stream>>>(single, ln_s_g, ln_s_b, s16);

  // 2) weights f32 -> f16, pre-swizzled into B-fragment layout
  int nW = (int)WW;
  int pkBlocks = (nW + 255) / 256;
  pack_w_kernel<<<pkBlocks, 256, 0, stream>>>(Wq,    WqP, CS, CS);
  pack_w_kernel<<<pkBlocks, 256, 0, stream>>>(Wk,    WkP, CS, CS);
  pack_w_kernel<<<pkBlocks, 256, 0, stream>>>(Wv,    WvP, CS, CS);
  pack_w_kernel<<<pkBlocks, 256, 0, stream>>>(Wgate, WgP, CS, CS);
  pack_w_kernel<<<pkBlocks, 256, 0, stream>>>(Wout,  WoP, CS, CS);

  // 3) projections
  const int gemmBlocks = (LSEQ / 16) * (CS / 64) / 4;
  gemm_kernel<<<gemmBlocks, 128, 0, stream>>>(s16, WqP, nullptr, nullptr, q16,
                                              LSEQ, CS, CS, 0.125f);
  gemm_kernel<<<gemmBlocks, 128, 0, stream>>>(s16, WkP, nullptr, nullptr, k16,
                                              LSEQ, CS, CS, 1.0f);
  gemm_kernel<<<gemmBlocks, 128, 0, stream>>>(s16, WvP, nullptr, nullptr, v16,
                                              LSEQ, CS, CS, 1.0f);
  gemm_kernel<<<gemmBlocks, 128, 0, stream>>>(s16, WgP, bgate, gate, nullptr,
                                              LSEQ, CS, CS, 1.0f);

  // 3b) pack v into B-fragment layout
  int nS = (int)SC;
  pack_h_kernel<<<(nS + 255) / 256, 256, 0, stream>>>(v16, vpack, LSEQ, CS);

  // 4) fused pair LN + z@Wb -> bias16 (double-buffered TDM). 16384 waves x 4 tiles.
  pair_bias_kernel<<<2048, 256, 0, stream>>>(pair, ln_z_g, ln_z_b, Wb, bias16);

  // 5) flash attention
  attn_kernel<<<192, 128, 0, stream>>>(q16, k16, vpack, bias16, o16);

  // 6) output projection
  gemm_kernel<<<gemmBlocks, 128, 0, stream>>>(o16, WoP, bout, oproj, nullptr,
                                              LSEQ, CS, CS, 1.0f);

  // 7) out = single + sigmoid(gate) * oproj
  final_kernel<<<(int)((SC + 255) / 256), 256, 0, stream>>>(single, gate, oproj, out, (int)SC);
}
